// CrossViewAttention_23132693856512
// MI455X (gfx1250) — compile-verified, matrix operand order checked by reading, not run on hardware
//
#include <hip/hip_runtime.h>
#include <hip/hip_fp16.h>

typedef __attribute__((ext_vector_type(16))) _Float16 v16h;
typedef __attribute__((ext_vector_type(8)))  float    v8f;
typedef __attribute__((ext_vector_type(4)))  int      v4i;

#define D_MODEL 768
#define N_VIEWS 4
#define N_PATCH 784
#define SEQ_T   3136            // V*P
#define M_ROWS  100352          // B*P*V
#define QKV_N   2304
#define FFN_N   3072

// CDNA5 async global->LDS copy path (ASYNCcnt) if the toolchain exposes it.
#if __has_builtin(__builtin_amdgcn_global_load_async_to_lds_b128) && \
    __has_builtin(__builtin_amdgcn_s_wait_asynccnt)
#define USE_ASYNC_LDS 1
#else
#define USE_ASYNC_LDS 0
#endif

__device__ __forceinline__ void cp16_g2l(const _Float16* g, _Float16* l) {
#if USE_ASYNC_LDS
    __builtin_amdgcn_global_load_async_to_lds_b128(
        (__attribute__((address_space(1))) v4i*)g,
        (__attribute__((address_space(3))) v4i*)l, 0, 0);
#else
    *(uint4*)l = *(const uint4*)g;
#endif
}

__device__ __forceinline__ void stage_drain() {
#if USE_ASYNC_LDS
    __builtin_amdgcn_s_wait_asynccnt(0);
#endif
}

// ---------------------------------------------------------------------------
// fp32 -> fp16 weight conversion (one-time, tiny)
// ---------------------------------------------------------------------------
__global__ void cvt_f32_f16(const float* __restrict__ s, _Float16* __restrict__ d, int n) {
    int i = blockIdx.x * 256 + threadIdx.x;
    if (i < n) d[i] = (_Float16)s[i];
}

// ---------------------------------------------------------------------------
// LayerNorm with view-fold remap: reads token layout (B,T,D), writes folded
// row layout r = (b*P+p)*V + v  as fp16 (WMMA GEMM A operand).
// ---------------------------------------------------------------------------
__global__ void ln_f16_kernel(const float* __restrict__ src,
                              const float* __restrict__ w,
                              const float* __restrict__ b,
                              _Float16* __restrict__ dst) {
    const int r  = blockIdx.x;
    const int bp = r >> 2, v = r & 3;
    const int bb = bp / N_PATCH, p = bp % N_PATCH;
    const size_t srow = (size_t)bb * SEQ_T + (size_t)v * N_PATCH + p;
    const float* xr = src + srow * D_MODEL;

    const int t = threadIdx.x;
    float x0 = xr[t], x1 = xr[t + 256], x2 = xr[t + 512];

    __shared__ float s1[256], s2[256];
    s1[t] = x0 + x1 + x2;
    s2[t] = x0 * x0 + x1 * x1 + x2 * x2;
    __syncthreads();
    for (int off = 128; off > 0; off >>= 1) {
        if (t < off) { s1[t] += s1[t + off]; s2[t] += s2[t + off]; }
        __syncthreads();
    }
    const float mu  = s1[0] * (1.0f / 768.0f);
    const float var = s2[0] * (1.0f / 768.0f) - mu * mu;
    const float rs  = rsqrtf(var + 1e-5f);

    _Float16* dr = dst + (size_t)r * D_MODEL;
    dr[t]       = (_Float16)((x0 - mu) * rs * w[t]       + b[t]);
    dr[t + 256] = (_Float16)((x1 - mu) * rs * w[t + 256] + b[t + 256]);
    dr[t + 512] = (_Float16)((x2 - mu) * rs * w[t + 512] + b[t + 512]);
}

// ---------------------------------------------------------------------------
// WMMA GEMM:  C[M,N] = A[M,K] (f16) @ W[N,K]^T (f16) + bias, fused epilogue.
//   FUSE 0: store f16 | FUSE 1: exact GELU, f16 | FUSE 2: residual + un-fold, f32
// 128x128 block tile, BK=64, double-buffered LDS (2x32KB), 256 thr = 8 wave32
// in a 4x2 grid; each wave computes 32x64 of C -> 16 WMMAs per K-stage.
// ---------------------------------------------------------------------------
template <int FUSE>
__global__ void gemm_wmma_kernel(const _Float16* __restrict__ A,
                                 const _Float16* __restrict__ W,
                                 const float*    __restrict__ bias,
                                 _Float16*       __restrict__ C16,
                                 float*          __restrict__ Cres,
                                 const float*    __restrict__ Res,
                                 int N, int K) {
    __shared__ __align__(16) _Float16 As[2][128 * 64];   // 32 KB
    __shared__ __align__(16) _Float16 Bs[2][128 * 64];   // 32 KB

    const int tid  = threadIdx.x;
    const int m0   = blockIdx.y * 128;
    const int n0   = blockIdx.x * 128;
    const int wid  = tid >> 5, lane = tid & 31;
    const int mw   = wid >> 1, nw = wid & 1;             // 4 x 2 wave grid
    const int lrow = lane & 15;
    const int ksel = (lane >> 4) * 8;                    // ISA 16b A/B layout
    const int lm   = (lane >> 4) * 8;                    // C row group

    // staging: 128 rows x 64 halves per tile; thread -> 4 chunks of 16B each
    const int srow = tid >> 3;                           // 0..31
    const int skof = (tid & 7) * 8;                      // 0..56

    auto stage = [&](int buf, int kb) {
        #pragma unroll
        for (int i = 0; i < 4; ++i) {
            const int row = srow + i * 32;
            cp16_g2l(A + (size_t)(m0 + row) * K + kb + skof, &As[buf][row * 64 + skof]);
            cp16_g2l(W + (size_t)(n0 + row) * K + kb + skof, &Bs[buf][row * 64 + skof]);
        }
    };

    v8f acc[2][4] = {};

    int cur = 0;
    stage(0, 0);
    stage_drain();
    __syncthreads();

    for (int kb = 0; kb < K; kb += 64) {
        const bool more = (kb + 64) < K;
        if (more) stage(cur ^ 1, kb + 64);               // overlaps WMMA burst

        #pragma unroll
        for (int ks = 0; ks < 2; ++ks) {
            union { v16h h; uint4 u[2]; } fa0, fa1;
            const _Float16* ab = &As[cur][(mw * 32 + lrow) * 64 + ks * 32 + ksel];
            fa0.u[0] = *(const uint4*)(ab);
            fa0.u[1] = *(const uint4*)(ab + 16);
            fa1.u[0] = *(const uint4*)(ab + 16 * 64);
            fa1.u[1] = *(const uint4*)(ab + 16 * 64 + 16);
            #pragma unroll
            for (int nt = 0; nt < 4; ++nt) {
                union { v16h h; uint4 u[2]; } fb;
                const _Float16* bb = &Bs[cur][(nw * 64 + nt * 16 + lrow) * 64 + ks * 32 + ksel];
                fb.u[0] = *(const uint4*)(bb);
                fb.u[1] = *(const uint4*)(bb + 16);
                acc[0][nt] = __builtin_amdgcn_wmma_f32_16x16x32_f16(
                    false, fa0.h, false, fb.h, (short)0, acc[0][nt], false, false);
                acc[1][nt] = __builtin_amdgcn_wmma_f32_16x16x32_f16(
                    false, fa1.h, false, fb.h, (short)0, acc[1][nt], false, false);
            }
        }

        stage_drain();
        __syncthreads();
        cur ^= 1;
    }

    // epilogue (ISA 32b C layout: VGPR j -> row, lane -> col)
    #pragma unroll
    for (int mt = 0; mt < 2; ++mt) {
        #pragma unroll
        for (int j = 0; j < 8; ++j) {
            const int gm = m0 + mw * 32 + mt * 16 + lm + j;
            size_t orow = 0;
            if constexpr (FUSE == 2) {
                const int bp = gm >> 2, vv = gm & 3;
                const int bb = bp / N_PATCH, p = bp % N_PATCH;
                orow = ((size_t)bb * SEQ_T + (size_t)vv * N_PATCH + p) * D_MODEL;
            }
            #pragma unroll
            for (int nt = 0; nt < 4; ++nt) {
                const int gn = n0 + nw * 64 + nt * 16 + lrow;
                float v = acc[mt][nt][j] + bias[gn];
                if constexpr (FUSE == 1) {
                    v = 0.5f * v * (1.0f + erff(v * 0.7071067811865475f));
                }
                if constexpr (FUSE == 2) {
                    Cres[orow + gn] = v + Res[orow + gn];
                } else {
                    C16[(size_t)gm * N + gn] = (_Float16)v;
                }
            }
        }
    }
}

// ---------------------------------------------------------------------------
// Cross-view attention: seq-len V=4, H=8 heads of dh=96.
// One block per (b,p); one wave per head; lane owns 3 of 96 dims.
// ---------------------------------------------------------------------------
__global__ void attn_kernel(const _Float16* __restrict__ qkv,
                            _Float16* __restrict__ o) {
    const int bp   = blockIdx.x;
    const int h    = threadIdx.x >> 5;
    const int lane = threadIdx.x & 31;
    const int c0   = lane * 3;

    float q[4][3], k[4][3], vv[4][3];
    const size_t rowbase = (size_t)bp * 4 * QKV_N;
    #pragma unroll
    for (int i = 0; i < 4; ++i) {
        const _Float16* r = qkv + rowbase + (size_t)i * QKV_N + h * 96 + c0;
        #pragma unroll
        for (int c = 0; c < 3; ++c) {
            q[i][c]  = (float)r[c];
            k[i][c]  = (float)r[768 + c];
            vv[i][c] = (float)r[1536 + c];
        }
    }

    float s[4][4];
    #pragma unroll
    for (int i = 0; i < 4; ++i)
        #pragma unroll
        for (int j = 0; j < 4; ++j) {
            float p = q[i][0] * k[j][0] + q[i][1] * k[j][1] + q[i][2] * k[j][2];
            for (int off = 16; off; off >>= 1) p += __shfl_xor(p, off, 32);
            s[i][j] = p * 0.10206207261596575f;   // 1/sqrt(96)
        }

    #pragma unroll
    for (int i = 0; i < 4; ++i) {
        const float m = fmaxf(fmaxf(s[i][0], s[i][1]), fmaxf(s[i][2], s[i][3]));
        float e[4], sum = 0.0f;
        #pragma unroll
        for (int j = 0; j < 4; ++j) { e[j] = __expf(s[i][j] - m); sum += e[j]; }
        const float inv = 1.0f / sum;
        float oa[3] = {0.0f, 0.0f, 0.0f};
        #pragma unroll
        for (int j = 0; j < 4; ++j) {
            const float pj = e[j] * inv;
            oa[0] += pj * vv[j][0]; oa[1] += pj * vv[j][1]; oa[2] += pj * vv[j][2];
        }
        _Float16* orow = o + (size_t)(bp * 4 + i) * D_MODEL + h * 96 + c0;
        orow[0] = (_Float16)oa[0];
        orow[1] = (_Float16)oa[1];
        orow[2] = (_Float16)oa[2];
    }
}

// ---------------------------------------------------------------------------
// Host launcher
// ---------------------------------------------------------------------------
extern "C" void kernel_launch(void* const* d_in, const int* in_sizes, int n_in,
                              void* d_out, int out_size, void* d_ws, size_t ws_size,
                              hipStream_t stream) {
    (void)in_sizes; (void)n_in; (void)out_size; (void)ws_size;

    const float* x    = (const float*)d_in[0];
    const float* n1w  = (const float*)d_in[2];
    const float* n1b  = (const float*)d_in[3];
    const float* ipw  = (const float*)d_in[4];
    const float* ipb  = (const float*)d_in[5];
    const float* ow   = (const float*)d_in[6];
    const float* ob   = (const float*)d_in[7];
    const float* n2w  = (const float*)d_in[8];
    const float* n2b  = (const float*)d_in[9];
    const float* f1w  = (const float*)d_in[10];
    const float* f1b  = (const float*)d_in[11];
    const float* f2w  = (const float*)d_in[12];
    const float* f2b  = (const float*)d_in[13];
    float* out = (float*)d_out;

    // workspace layout (bytes)
    const size_t OFF_QKV = (size_t)M_ROWS * D_MODEL * 2;
    const size_t OFF_ACT = OFF_QKV + (size_t)M_ROWS * QKV_N * 2;
    const size_t OFF_W   = OFF_ACT + (size_t)M_ROWS * FFN_N * 2;
    char* ws = (char*)d_ws;
    _Float16* h16   = (_Float16*)(ws);            // h1, later reused for attn output
    _Float16* qkv16 = (_Float16*)(ws + OFF_QKV);  // qkv, later reused for h2
    _Float16* act16 = (_Float16*)(ws + OFF_ACT);  // ffn hidden
    _Float16* w_ip  = (_Float16*)(ws + OFF_W);
    _Float16* w_o   = w_ip + (size_t)QKV_N * D_MODEL;
    _Float16* w_f1  = w_o  + (size_t)D_MODEL * D_MODEL;
    _Float16* w_f2  = w_f1 + (size_t)FFN_N * D_MODEL;
    _Float16* o16   = h16;
    _Float16* h2_16 = qkv16;

    // 1) weight conversion fp32 -> fp16
    cvt_f32_f16<<<(QKV_N * D_MODEL + 255) / 256, 256, 0, stream>>>(ipw, w_ip, QKV_N * D_MODEL);
    cvt_f32_f16<<<(D_MODEL * D_MODEL + 255) / 256, 256, 0, stream>>>(ow, w_o, D_MODEL * D_MODEL);
    cvt_f32_f16<<<(FFN_N * D_MODEL + 255) / 256, 256, 0, stream>>>(f1w, w_f1, FFN_N * D_MODEL);
    cvt_f32_f16<<<(D_MODEL * FFN_N + 255) / 256, 256, 0, stream>>>(f2w, w_f2, D_MODEL * FFN_N);

    // 2) LN1 (fold remap) -> h1 f16
    ln_f16_kernel<<<M_ROWS, 256, 0, stream>>>(x, n1w, n1b, h16);

    // 3) QKV = h1 @ Wip^T + b  (f16 out)
    gemm_wmma_kernel<0><<<dim3(QKV_N / 128, M_ROWS / 128), 256, 0, stream>>>(
        h16, w_ip, ipb, qkv16, nullptr, nullptr, QKV_N, D_MODEL);

    // 4) tiny cross-view attention -> o16
    attn_kernel<<<M_ROWS / 4, 256, 0, stream>>>(qkv16, o16);

    // 5) out-proj + residual(x), un-fold into d_out (f32)
    gemm_wmma_kernel<2><<<dim3(D_MODEL / 128, M_ROWS / 128), 256, 0, stream>>>(
        o16, w_o, ob, nullptr, out, x, D_MODEL, D_MODEL);

    // 6) LN2 reads xv from d_out -> h2 f16
    ln_f16_kernel<<<M_ROWS, 256, 0, stream>>>(out, n2w, n2b, h2_16);

    // 7) FFN1 + exact GELU (f16 out)
    gemm_wmma_kernel<1><<<dim3(FFN_N / 128, M_ROWS / 128), 256, 0, stream>>>(
        h2_16, w_f1, f1b, act16, nullptr, nullptr, FFN_N, D_MODEL);

    // 8) FFN2 + residual(d_out) -> d_out (f32)
    gemm_wmma_kernel<2><<<dim3(D_MODEL / 128, M_ROWS / 128), 256, 0, stream>>>(
        act16, w_f2, f2b, nullptr, out, out, D_MODEL, FFN_N);
}
